// MambaSlotAttention_25563645346565
// MI455X (gfx1250) — matrix operand order repaired and checked by reading, not compiled, not run on hardware
//
#include <hip/hip_runtime.h>
#include <cstdint>
#include <cstddef>

#define DEV __device__ __forceinline__

typedef __attribute__((ext_vector_type(16))) __bf16        v16bf;
typedef __attribute__((ext_vector_type(8)))  float         v8f;
typedef __attribute__((ext_vector_type(4)))  unsigned int  u32x4;
typedef __attribute__((ext_vector_type(8)))  unsigned int  u32x8;

namespace cfg {
constexpr int Db = 768, Kslots = 12, Titer = 3, Bb = 32, Nf = 1369;
constexpr int NP = 1408;                 // N padded to multiple of 64
constexpr int H = 8, HD = 96, D3 = 2304, D4 = 3072;
constexpr int Mfeat = Bb * Nf;           // 43808 = 16 * 2738
constexpr int Mslot = Bb * Kslots;       // 384   = 16 * 24
constexpr float SCALE  = 0.03608439182435161f;  // 768^-0.5
constexpr float HSCALE = 0.10206207261596575f;  // 96^-0.5
}

typedef unsigned short bfr;  // raw bf16 storage

union FragU {
  v16bf v;
  u32x4 q[2];
  unsigned short u[16];
};

DEV float bf2f(unsigned short h) {
  union { unsigned int u; float f; } c; c.u = ((unsigned int)h) << 16; return c.f;
}
DEV unsigned short f2bf(float f) {
  union { float f; unsigned int u; } c; c.f = f;
  unsigned int u = c.u;
  unsigned int r = u + 0x7FFFu + ((u >> 16) & 1u);  // round-to-nearest-even
  return (unsigned short)(r >> 16);
}
DEV float gelu_exact(float x) { return 0.5f * x * (1.0f + erff(x * 0.7071067811865476f)); }

// ---- Unguarded WMMA fragment loaders (bf16, 16x16x32) --------------------
DEV v16bf load_fragA(const bfr* __restrict__ rowptr, int kb, int lane) {
  FragU f;
  int k0 = kb + ((lane & 16) ? 16 : 0);
  const u32x4* p = (const u32x4*)(rowptr + k0);
  f.q[0] = p[0];       // K = k0 .. k0+7
  f.q[1] = p[1];       // K = k0+8 .. k0+15
  return f.v;
}
DEV v16bf load_fragB(const bfr* __restrict__ rowptr, int kb, int lane) {
  FragU f;
  int k0 = kb + ((lane & 16) ? 16 : 0);
  const u32x4* p = (const u32x4*)(rowptr + k0);
  f.q[0] = p[0];       // K = k0 .. k0+7
  f.q[1] = p[1];       // K = k0+8 .. k0+15
  return f.v;
}
DEV v8f wmma_bf16(v16bf a, v16bf b, v8f c) {
  return __builtin_amdgcn_wmma_f32_16x16x32_bf16(false, a, false, b, (short)0, c, false, false);
}
DEV v8f zero8() { v8f z;
#pragma unroll
  for (int i = 0; i < 8; i++) z[i] = 0.f; return z; }

// ---- LayerNorm: x (+optional x2) -> bf16 and/or fp32, D=768 --------------
__global__ __launch_bounds__(256) void ln_kernel(
    const float* __restrict__ x, const float* __restrict__ x2,
    const float* __restrict__ g, const float* __restrict__ be,
    bfr* __restrict__ out_bf, float* __restrict__ out_f, int rows)
{
  int row = blockIdx.x;
  if (row >= rows) return;
  const float* xr  = x + (size_t)row * cfg::Db;
  const float* x2r = x2 ? x2 + (size_t)row * cfg::Db : nullptr;
  float v[3]; float s = 0.f, s2 = 0.f;
#pragma unroll
  for (int i = 0; i < 3; i++) {
    int e = threadIdx.x + i * 256;
    float t = xr[e];
    if (x2r) t += x2r[e];
    v[i] = t; s += t; s2 += t * t;
  }
  __shared__ float rs[256], rq[256];
  rs[threadIdx.x] = s; rq[threadIdx.x] = s2;
  __syncthreads();
  for (int o = 128; o > 0; o >>= 1) {
    if (threadIdx.x < o) { rs[threadIdx.x] += rs[threadIdx.x + o]; rq[threadIdx.x] += rq[threadIdx.x + o]; }
    __syncthreads();
  }
  float mean = rs[0] * (1.f / 768.f);
  float var  = rq[0] * (1.f / 768.f) - mean * mean;
  float rstd = rsqrtf(var + 1e-5f);
#pragma unroll
  for (int i = 0; i < 3; i++) {
    int e = threadIdx.x + i * 256;
    float t = (v[i] - mean) * rstd * g[e] + be[e];
    if (out_bf) out_bf[(size_t)row * cfg::Db + e] = f2bf(t);
    if (out_f)  out_f [(size_t)row * cfg::Db + e] = t;
  }
}

// ---- fp32 [Kd,Nc] -> bf16 transposed [Nc,Kd] -----------------------------
__global__ __launch_bounds__(256) void transpose_to_bf16(
    const float* __restrict__ W, bfr* __restrict__ WT, int Kd, int Nc)
{
  __shared__ float tile[32][33];
  int kt = blockIdx.y * 32, nt = blockIdx.x * 32;
#pragma unroll
  for (int i = 0; i < 4; i++) {
    int r = kt + threadIdx.y + i * 8, c = nt + threadIdx.x;
    tile[threadIdx.y + i * 8][threadIdx.x] = (r < Kd && c < Nc) ? W[(size_t)r * Nc + c] : 0.f;
  }
  __syncthreads();
#pragma unroll
  for (int i = 0; i < 4; i++) {
    int n = nt + threadIdx.y + i * 8, k0 = kt + threadIdx.x;
    if (n < Nc && k0 < Kd) WT[(size_t)n * Kd + k0] = f2bf(tile[threadIdx.x][threadIdx.y + i * 8]);
  }
}

// ---- Fused K/V projection over features ----------------------------------
// Waves in a block share tn (same weight tiles -> WGP$ hits); tm varies.
__global__ __launch_bounds__(256) void kv_kernel(
    const bfr* __restrict__ fn, const bfr* __restrict__ WkT, const bfr* __restrict__ WvT,
    bfr* __restrict__ kp, bfr* __restrict__ vp)
{
  using namespace cfg;
  constexpr int tiles_n = Db / 64;                   // 12
  constexpr int tiles_m = Mfeat / 16;                // 2738
  int wid = (int)((blockIdx.x * blockDim.x + threadIdx.x) >> 5);
  if (wid >= tiles_m * tiles_n) return;
  int tm = wid % tiles_m, tn = wid / tiles_m;
  int lane = threadIdx.x & 31;
  int lrow = lane & 15;
  const bfr* arow = fn + (size_t)(tm * 16 + lrow) * Db;
  const bfr* bk0  = WkT + (size_t)(tn * 64 + lrow) * Db;
  const bfr* bv0  = WvT + (size_t)(tn * 64 + lrow) * Db;
  v8f ak[4], av[4];
#pragma unroll
  for (int s = 0; s < 4; s++) { ak[s] = zero8(); av[s] = zero8(); }
  for (int kb = 0; kb < Db; kb += 32) {
    __builtin_prefetch((const void*)(arow + kb + 64), 0, 1);
    v16bf a = load_fragA(arow, kb, lane);
#pragma unroll
    for (int s = 0; s < 4; s++) {
      v16bf b = load_fragB(bk0 + (size_t)s * 16 * Db, kb, lane);
      ak[s] = wmma_bf16(a, b, ak[s]);
    }
#pragma unroll
    for (int s = 0; s < 4; s++) {
      v16bf b = load_fragB(bv0 + (size_t)s * 16 * Db, kb, lane);
      av[s] = wmma_bf16(a, b, av[s]);
    }
  }
  int rsel = (lane >> 4) & 1;
#pragma unroll
  for (int r = 0; r < 8; r++) {
    int m = tm * 16 + rsel * 8 + r;
    int b = m / Nf;                   // batch
    int ni = m - b * Nf;              // row within batch
    size_t orow = ((size_t)b * NP + ni) * Db;
#pragma unroll
    for (int s = 0; s < 4; s++) {
      int n = tn * 64 + s * 16 + lrow;
      kp[orow + n] = f2bf(ak[s][r]);
      vp[orow + n] = f2bf(av[s][r]);
    }
  }
}

// ---- Generic GEMM (exact tiles): C[M,N] = A[M,Kd] * BT[N,Kd]^T -----------
template<bool OUT_BF16, bool HAS_BIAS, bool DO_GELU, bool HAS_RES>
__global__ __launch_bounds__(256) void gemm_kernel(
    const bfr* __restrict__ A, const bfr* __restrict__ BT,
    const float* __restrict__ bias, const float* __restrict__ res,
    void* __restrict__ out, int M, int N, int Kd)
{
  int tiles_n = N >> 6;
  int tiles_m = M >> 4;
  int wid = (int)((blockIdx.x * blockDim.x + threadIdx.x) >> 5);
  if (wid >= tiles_m * tiles_n) return;
  int tm = wid % tiles_m, tn = wid / tiles_m;      // tm fast -> B reuse in block
  int lane = threadIdx.x & 31;
  int lrow = lane & 15;
  const bfr* arow = A  + (size_t)(tm * 16 + lrow) * Kd;
  const bfr* brow = BT + (size_t)(tn * 64 + lrow) * Kd;
  v8f acc[4];
#pragma unroll
  for (int s = 0; s < 4; s++) acc[s] = zero8();
  for (int kb = 0; kb < Kd; kb += 32) {
    v16bf a = load_fragA(arow, kb, lane);
#pragma unroll
    for (int s = 0; s < 4; s++) {
      v16bf b = load_fragB(brow + (size_t)s * 16 * Kd, kb, lane);
      acc[s] = wmma_bf16(a, b, acc[s]);
    }
  }
  int rsel = (lane >> 4) & 1;
#pragma unroll
  for (int s = 0; s < 4; s++) {
    int n = tn * 64 + s * 16 + lrow;
    float bv = HAS_BIAS ? bias[n] : 0.f;
#pragma unroll
    for (int r = 0; r < 8; r++) {
      int m = tm * 16 + rsel * 8 + r;
      float vv = acc[s][r] + bv;
      if (HAS_RES) vv += res[(size_t)m * N + n];
      if (DO_GELU) vv = gelu_exact(vv);
      if (OUT_BF16) ((bfr*)out)[(size_t)m * N + n] = f2bf(vv);
      else          ((float*)out)[(size_t)m * N + n] = vv;
    }
  }
}

// ---- zero pad rows of kp/vp ([B,NP,Db]) and attn ([B,16,NP]) -------------
__global__ __launch_bounds__(256) void pad_zero_kernel(
    bfr* __restrict__ kp, bfr* __restrict__ vp, bfr* __restrict__ attn)
{
  using namespace cfg;
  int t = blockIdx.x * 256 + threadIdx.x;
  const int padKV = Bb * (NP - Nf) * Db;
  if (t < padKV) {
    int b = t / ((NP - Nf) * Db);
    int r = t - b * ((NP - Nf) * Db);
    int row = r / Db, c = r - row * Db;
    size_t idx = ((size_t)b * NP + Nf + row) * Db + c;
    kp[idx] = 0; vp[idx] = 0;
  }
  const int padA = Bb * 4 * NP;
  if (t < padA) {
    int b = t / (4 * NP);
    int r = t - b * (4 * NP);
    int row = r / NP, c = r - row * NP;
    attn[((size_t)b * 16 + 12 + row) * NP + c] = 0;
  }
}

// ---- q [B*12,Db] -> padded qp [B,16,Db] (rows 12..15 zero) ---------------
__global__ __launch_bounds__(256) void qpad_kernel(const bfr* __restrict__ qb, bfr* __restrict__ qp)
{
  using namespace cfg;
  int t = blockIdx.x * 256 + threadIdx.x;
  if (t >= Bb * 16 * Db) return;
  int b = t / (16 * Db);
  int r = t - b * (16 * Db);
  int row = r / Db, c = r - row * Db;
  qp[t] = (row < Kslots) ? qb[((size_t)b * Kslots + row) * Db + c] : (bfr)0;
}

// ---- Batched NT GEMM: logits[b,k,n] = SCALE * q[b,k,:] . kmat[b,n,:] -----
__global__ __launch_bounds__(256) void logits_kernel(
    const bfr* __restrict__ qp, const bfr* __restrict__ kp, float* __restrict__ logits)
{
  using namespace cfg;
  int b = blockIdx.y;
  int wid = (int)((blockIdx.x * blockDim.x + threadIdx.x) >> 5);
  constexpr int tiles_n = NP / 64;           // 22
  if (wid >= tiles_n) return;
  int lane = threadIdx.x & 31;
  int lrow = lane & 15;
  const bfr* arow = qp + (size_t)b * 16 * Db + (size_t)lrow * Db;
  const bfr* brow = kp + (size_t)b * NP * Db + (size_t)(wid * 64 + lrow) * Db;
  v8f acc[4];
#pragma unroll
  for (int s = 0; s < 4; s++) acc[s] = zero8();
  for (int kb = 0; kb < Db; kb += 32) {
    v16bf a = load_fragA(arow, kb, lane);
#pragma unroll
    for (int s = 0; s < 4; s++) {
      v16bf bb = load_fragB(brow + (size_t)s * 16 * Db, kb, lane);
      acc[s] = wmma_bf16(a, bb, acc[s]);
    }
  }
  int rsel = (lane >> 4) & 1;
#pragma unroll
  for (int s = 0; s < 4; s++) {
    int n = wid * 64 + s * 16 + lrow;
    if (n >= Nf) continue;
#pragma unroll
    for (int r = 0; r < 8; r++) {
      int m = rsel * 8 + r;
      if (m < Kslots)
        logits[((size_t)b * Kslots + m) * NP + n] = acc[s][r] * SCALE;
    }
  }
}

// ---- softmax over slots (k) for each (b,n); also writes final masks ------
template<bool FINAL>
__global__ __launch_bounds__(256) void softmax_kernel(
    const float* __restrict__ logits, bfr* __restrict__ attn, float* __restrict__ masks)
{
  using namespace cfg;
  int t = blockIdx.x * 256 + threadIdx.x;
  if (t >= Bb * NP) return;
  int b = t / NP, n = t - b * NP;
  size_t abase = (size_t)b * 16 * NP + n;
  if (n >= Nf) {
#pragma unroll
    for (int k = 0; k < Kslots; k++) attn[abase + (size_t)k * NP] = 0;
    return;
  }
  size_t lbase = (size_t)b * Kslots * NP + n;
  float l[Kslots]; float mx = -3.0e38f;
#pragma unroll
  for (int k = 0; k < Kslots; k++) { l[k] = logits[lbase + (size_t)k * NP]; mx = fmaxf(mx, l[k]); }
  float sum = 0.f;
#pragma unroll
  for (int k = 0; k < Kslots; k++) { l[k] = expf(l[k] - mx); sum += l[k]; }
  float inv = 1.f / sum;
#pragma unroll
  for (int k = 0; k < Kslots; k++) {
    float a = l[k] * inv;
    attn[abase + (size_t)k * NP] = f2bf(a);
    if (FINAL) masks[((size_t)b * Nf + n) * Kslots + k] = a;
  }
}

// ---- mass[b,k] = 1 / max(sum_n attn[b,k,n], EPS) -------------------------
// Row is DMA'd into LDS by the Tensor Data Mover (wave 0 issues descriptor),
// then the whole block reduces out of LDS.
__global__ __launch_bounds__(256) void mass_kernel(const bfr* __restrict__ attn,
                                                   float* __restrict__ massr)
{
  using namespace cfg;
  __shared__ unsigned short lbuf[NP];
  __shared__ float red[256];
  int blk = blockIdx.x;                  // b*12 + k
  int b = blk / Kslots, k = blk - b * Kslots;
  const bfr* row = attn + ((size_t)b * 16 + k) * NP;
  if (threadIdx.x < 32) {
    unsigned ldsa = (unsigned)(uintptr_t)(void*)&lbuf[0];
    unsigned long long ga = (unsigned long long)(uintptr_t)row;
    u32x4 g0;
    g0[0] = 1u;                                   // count=1 (valid user D#)
    g0[1] = ldsa;                                 // lds_addr
    g0[2] = (unsigned)(ga & 0xFFFFFFFFu);         // global_addr[31:0]
    g0[3] = (unsigned)((ga >> 32) & 0x01FFFFFFu)  // global_addr[56:32]
            | (2u << 30);                         // type = 2 (image)
    u32x8 g1;
    g1[0] = 0x00010000u;                          // data_size=1 (2B), mask=0
    g1[1] = (unsigned)(NP & 0xFFFF) << 16;        // tensor_dim0[15:0] @ bits 63:48
    g1[2] = 0x00010000u;                          // tensor_dim0 hi=0, tensor_dim1=1
    g1[3] = (unsigned)(NP & 0xFFFF) << 16;        // tile_dim0 = NP
    g1[4] = 1u;                                   // tile_dim1 = 1
    g1[5] = (unsigned)NP;                         // tensor_dim0_stride (elements)
    g1[6] = 0u;
    g1[7] = 0u;
    asm volatile("tensor_load_to_lds %0, %1" :: "s"(g0), "s"(g1) : "memory");
    __builtin_amdgcn_s_wait_tensorcnt(0);
  }
  __syncthreads();
  float s = 0.f;
  for (int i = threadIdx.x; i < NP; i += 256) s += bf2f(lbuf[i]);
  red[threadIdx.x] = s; __syncthreads();
  for (int o = 128; o > 0; o >>= 1) {
    if (threadIdx.x < o) red[threadIdx.x] += red[threadIdx.x + o];
    __syncthreads();
  }
  if (threadIdx.x == 0) massr[blk] = 1.f / fmaxf(red[0], 1e-8f);
}

// ---- updates[b,k,d] = (attn[b,k,:] . v[b,:,d]) * massr[b,k] --------------
__global__ __launch_bounds__(256) void updates_kernel(
    const bfr* __restrict__ attn, const bfr* __restrict__ vp,
    const float* __restrict__ massr, float* __restrict__ updates)
{
  using namespace cfg;
  int b = blockIdx.y;
  int wid = (int)((blockIdx.x * blockDim.x + threadIdx.x) >> 5);
  if (wid >= Db / 64) return;
  int lane = threadIdx.x & 31;
  int lrow = lane & 15;
  int hs = (lane & 16) ? 16 : 0;
  const bfr* arow = attn + (size_t)b * 16 * NP + (size_t)lrow * NP;
  const bfr* Vb   = vp   + (size_t)b * NP * Db;
  v8f acc[4];
#pragma unroll
  for (int s = 0; s < 4; s++) acc[s] = zero8();
  for (int nb = 0; nb < NP; nb += 32) {
    v16bf a = load_fragA(arow, nb, lane);
#pragma unroll
    for (int s = 0; s < 4; s++) {
      int d = wid * 64 + s * 16 + lrow;
      FragU fb;
#pragma unroll
      for (int j = 0; j < 16; j++) {
        fb.u[j] = Vb[(size_t)(nb + hs + j) * Db + d];
      }
      acc[s] = wmma_bf16(a, fb.v, acc[s]);
    }
  }
  int rsel = (lane >> 4) & 1;
#pragma unroll
  for (int s = 0; s < 4; s++) {
    int d = wid * 64 + s * 16 + lrow;
#pragma unroll
    for (int r = 0; r < 8; r++) {
      int m = rsel * 8 + r;
      if (m < Kslots)
        updates[((size_t)b * Kslots + m) * Db + d] = acc[s][r] * massr[b * Kslots + m];
    }
  }
}

// ---- 12-slot, 8-head self-attention (per (b,h) block) --------------------
__global__ __launch_bounds__(160) void tiny_attn_kernel(
    const bfr* __restrict__ qkv, bfr* __restrict__ o)
{
  using namespace cfg;
  int bh = blockIdx.x; int b = bh / H; int h = bh - b * H;
  __shared__ float sc[Kslots][Kslots];
  const bfr* base = qkv + (size_t)b * Kslots * D3;
  int t = threadIdx.x;
  if (t < Kslots * Kslots) {
    int qi = t / Kslots, ki = t - qi * Kslots;
    const bfr* qp2 = base + (size_t)qi * D3 + h * HD;
    const bfr* kp2 = base + (size_t)ki * D3 + Db + h * HD;
    float s = 0.f;
    for (int d = 0; d < HD; d++) s += bf2f(qp2[d]) * bf2f(kp2[d]);
    sc[qi][ki] = s * HSCALE;
  }
  __syncthreads();
  if (t < Kslots) {
    float mx = -3.0e38f;
    for (int k = 0; k < Kslots; k++) mx = fmaxf(mx, sc[t][k]);
    float sum = 0.f;
    for (int k = 0; k < Kslots; k++) { float e = expf(sc[t][k] - mx); sc[t][k] = e; sum += e; }
    float inv = 1.f / sum;
    for (int k = 0; k < Kslots; k++) sc[t][k] *= inv;
  }
  __syncthreads();
  for (int idx = t; idx < Kslots * HD; idx += 160) {
    int qi = idx / HD, d = idx - qi * HD;
    float acc = 0.f;
    for (int ki = 0; ki < Kslots; ki++)
      acc += sc[qi][ki] * bf2f(base[(size_t)ki * D3 + 2 * Db + h * HD + d]);
    o[((size_t)b * Kslots + qi) * Db + h * HD + d] = f2bf(acc);
  }
}

// ==========================================================================
extern "C" void kernel_launch(void* const* d_in, const int* in_sizes, int n_in,
                              void* d_out, int out_size, void* d_ws, size_t ws_size,
                              hipStream_t stream)
{
  using namespace cfg;
  (void)in_sizes; (void)n_in; (void)out_size; (void)ws_size;
  const float* features   = (const float*)d_in[0];
  const float* slots_init = (const float*)d_in[1];
  const float* nf_g = (const float*)d_in[2],  *nf_b = (const float*)d_in[3];
  const float* ns_g = (const float*)d_in[4],  *ns_b = (const float*)d_in[5];
  const float* Wq   = (const float*)d_in[6],  *Wk   = (const float*)d_in[7];
  const float* Wv   = (const float*)d_in[8];
  const float* mg   = (const float*)d_in[9],  *mb   = (const float*)d_in[10];
  const float* mW1  = (const float*)d_in[11], *mb1  = (const float*)d_in[12];
  const float* mW2  = (const float*)d_in[13], *mb2  = (const float*)d_in[14];
  const float* b_ln1g = (const float*)d_in[15], *b_ln1b = (const float*)d_in[16];
  const float* b_Wqkv = (const float*)d_in[17], *b_bqkv = (const float*)d_in[18];
  const float* b_Wo   = (const float*)d_in[19], *b_bo   = (const float*)d_in[20];
  const float* b_ln2g = (const float*)d_in[21], *b_ln2b = (const float*)d_in[22];
  const float* b_W1   = (const float*)d_in[23], *b_b1   = (const float*)d_in[24];
  const float* b_W2   = (const float*)d_in[25], *b_b2   = (const float*)d_in[26];

  char* wsp = (char*)d_ws;
  size_t off = 0;
  auto alloc = [&](size_t bytes) -> void* {
    void* p = (void*)(wsp + off);
    off += (bytes + 255) & ~(size_t)255;
    return p;
  };
  bfr* fn   = (bfr*)alloc((size_t)Mfeat * Db * 2);
  bfr* kp   = (bfr*)alloc((size_t)Bb * NP * Db * 2);
  bfr* vp   = (bfr*)alloc((size_t)Bb * NP * Db * 2);
  bfr* WqT  = (bfr*)alloc((size_t)Db * Db * 2);
  bfr* WkT  = (bfr*)alloc((size_t)Db * Db * 2);
  bfr* WvT  = (bfr*)alloc((size_t)Db * Db * 2);
  bfr* mW1T = (bfr*)alloc((size_t)Db * D4 * 2);
  bfr* mW2T = (bfr*)alloc((size_t)D4 * Db * 2);
  bfr *WqkvT[Titer], *WoT[Titer], *W1T[Titer], *W2T[Titer];
  for (int t = 0; t < Titer; t++) {
    WqkvT[t] = (bfr*)alloc((size_t)D3 * Db * 2);
    WoT[t]   = (bfr*)alloc((size_t)Db * Db * 2);
    W1T[t]   = (bfr*)alloc((size_t)D4 * Db * 2);
    W2T[t]   = (bfr*)alloc((size_t)Db * D4 * 2);
  }
  bfr*   snb    = (bfr*)  alloc((size_t)Mslot * Db * 2);
  bfr*   qb     = (bfr*)  alloc((size_t)Mslot * Db * 2);
  bfr*   qpd    = (bfr*)  alloc((size_t)Bb * 16 * Db * 2);
  float* logits = (float*)alloc((size_t)Bb * Kslots * NP * 4);
  bfr*   attn   = (bfr*)  alloc((size_t)Bb * 16 * NP * 2);
  float* massr  = (float*)alloc((size_t)Mslot * 4);
  float* updates= (float*)alloc((size_t)Mslot * Db * 4);
  float* x0     = (float*)alloc((size_t)Mslot * Db * 4);
  float* x1     = (float*)alloc((size_t)Mslot * Db * 4);
  float* x2     = (float*)alloc((size_t)Mslot * Db * 4);
  bfr*   hbf    = (bfr*)  alloc((size_t)Mslot * Db * 2);
  bfr*   qkvb   = (bfr*)  alloc((size_t)Mslot * D3 * 2);
  bfr*   obf    = (bfr*)  alloc((size_t)Mslot * Db * 2);
  bfr*   h2bf   = (bfr*)  alloc((size_t)Mslot * Db * 2);
  bfr*   m1b    = (bfr*)  alloc((size_t)Mslot * D4 * 2);
  bfr*   hbb    = (bfr*)  alloc((size_t)Mslot * Db * 2);
  bfr*   m2b    = (bfr*)  alloc((size_t)Mslot * D4 * 2);
  float* slotsA = (float*)alloc((size_t)Mslot * Db * 4);
  float* slotsB = (float*)alloc((size_t)Mslot * Db * 4);

  dim3 tb(32, 8);
  auto transpose = [&](const float* W, bfr* WT, int Kd, int Nc) {
    dim3 g((Nc + 31) / 32, (Kd + 31) / 32);
    transpose_to_bf16<<<g, tb, 0, stream>>>(W, WT, Kd, Nc);
  };
  auto gemm_grid = [](int M, int N) {
    long waves = (long)(M >> 4) * (N >> 6);
    return dim3((unsigned)((waves * 32 + 255) / 256));
  };

  transpose(Wq, WqT, Db, Db);
  transpose(Wk, WkT, Db, Db);
  transpose(Wv, WvT, Db, Db);
  transpose(mW1, mW1T, Db, D4);
  transpose(mW2, mW2T, D4, Db);
  for (int t = 0; t < Titer; t++) {
    transpose(b_Wqkv + (size_t)t * Db * D3, WqkvT[t], Db, D3);
    transpose(b_Wo   + (size_t)t * Db * Db, WoT[t],   Db, Db);
    transpose(b_W1   + (size_t)t * Db * D4, W1T[t],   Db, D4);
    transpose(b_W2   + (size_t)t * D4 * Db, W2T[t],   D4, Db);
  }
  {
    int padmax = Bb * (NP - Nf) * Db;
    pad_zero_kernel<<<(padmax + 255) / 256, 256, 0, stream>>>(kp, vp, attn);
  }
  ln_kernel<<<Mfeat, 256, 0, stream>>>(features, nullptr, nf_g, nf_b, fn, nullptr, Mfeat);
  {
    long waves = (long)(Mfeat / 16) * (Db / 64);
    kv_kernel<<<(unsigned)((waves * 32 + 255) / 256), 256, 0, stream>>>(fn, WkT, WvT, kp, vp);
  }
  hipMemcpyAsync(slotsA, slots_init, (size_t)Mslot * Db * 4, hipMemcpyDeviceToDevice, stream);

  const int lg_blocks = ((NP / 64) * 32 + 255) / 256;
  const int up_blocks = ((Db / 64) * 32 + 255) / 256;
  const int sm_blocks = (Bb * NP + 255) / 256;
  const int qp_blocks = (Bb * 16 * Db + 255) / 256;

  for (int t = 0; t < Titer; t++) {
    ln_kernel<<<Mslot, 256, 0, stream>>>(slotsA, nullptr, ns_g, ns_b, snb, x0, Mslot);
    gemm_kernel<true,false,false,false><<<gemm_grid(Mslot, Db), 256, 0, stream>>>(
        snb, WqT, nullptr, nullptr, qb, Mslot, Db, Db);
    qpad_kernel<<<qp_blocks, 256, 0, stream>>>(qb, qpd);
    logits_kernel<<<dim3(lg_blocks, Bb), 256, 0, stream>>>(qpd, kp, logits);
    softmax_kernel<false><<<sm_blocks, 256, 0, stream>>>(logits, attn, nullptr);
    mass_kernel<<<Mslot, 256, 0, stream>>>(attn, massr);
    updates_kernel<<<dim3(up_blocks, Bb), 256, 0, stream>>>(attn, vp, massr, updates);
    ln_kernel<<<Mslot, 256, 0, stream>>>(x0, nullptr, b_ln1g + t * Db, b_ln1b + t * Db,
                                         hbf, nullptr, Mslot);
    gemm_kernel<true,true,false,false><<<gemm_grid(Mslot, D3), 256, 0, stream>>>(
        hbf, WqkvT[t], b_bqkv + (size_t)t * D3, nullptr, qkvb, Mslot, D3, Db);
    tiny_attn_kernel<<<Bb * H, 160, 0, stream>>>(qkvb, obf);
    gemm_kernel<false,true,false,true><<<gemm_grid(Mslot, Db), 256, 0, stream>>>(
        obf, WoT[t], b_bo + (size_t)t * Db, x0, x1, Mslot, Db, Db);
    ln_kernel<<<Mslot, 256, 0, stream>>>(x1, nullptr, b_ln2g + t * Db, b_ln2b + t * Db,
                                         h2bf, nullptr, Mslot);
    gemm_kernel<true,true,true,false><<<gemm_grid(Mslot, D4), 256, 0, stream>>>(
        h2bf, W1T[t], b_b1 + (size_t)t * D4, nullptr, m1b, Mslot, D4, Db);
    gemm_kernel<false,true,false,true><<<gemm_grid(Mslot, Db), 256, 0, stream>>>(
        m1b, W2T[t], b_b2 + (size_t)t * Db, x1, x2, Mslot, Db, D4);
    ln_kernel<<<Mslot, 256, 0, stream>>>(updates, x2, mg, mb, hbb, nullptr, Mslot);
    gemm_kernel<true,true,true,false><<<gemm_grid(Mslot, D4), 256, 0, stream>>>(
        hbb, mW1T, mb1, nullptr, m2b, Mslot, D4, Db);
    gemm_kernel<false,true,false,true><<<gemm_grid(Mslot, Db), 256, 0, stream>>>(
        m2b, mW2T, mb2, slotsA, slotsB, Mslot, Db, D4);
    float* tmp = slotsA; slotsA = slotsB; slotsB = tmp;
  }

  ln_kernel<<<Mslot, 256, 0, stream>>>(slotsA, nullptr, ns_g, ns_b, snb, nullptr, Mslot);
  gemm_kernel<true,false,false,false><<<gemm_grid(Mslot, Db), 256, 0, stream>>>(
      snb, WqT, nullptr, nullptr, qb, Mslot, Db, Db);
  qpad_kernel<<<qp_blocks, 256, 0, stream>>>(qb, qpd);
  logits_kernel<<<dim3(lg_blocks, Bb), 256, 0, stream>>>(qpd, kp, logits);
  float* masks_out = (float*)d_out + (size_t)Mslot * Db;
  softmax_kernel<true><<<sm_blocks, 256, 0, stream>>>(logits, attn, masks_out);
  hipMemcpyAsync(d_out, slotsA, (size_t)Mslot * Db * 4, hipMemcpyDeviceToDevice, stream);
}
